// GAT2_33105607917778
// MI455X (gfx1250) — matrix operand order repaired and checked by reading, not compiled, not run on hardware
//
#include <hip/hip_runtime.h>

// ---------------- types for WMMA / TDM (gfx1250, wave32) ----------------
typedef __attribute__((ext_vector_type(16))) __bf16        v16bf;
typedef __attribute__((ext_vector_type(8)))  float         v8f;
typedef __attribute__((ext_vector_type(4)))  unsigned int  v4u;
typedef __attribute__((ext_vector_type(8)))  unsigned int  v8u;
typedef __attribute__((ext_vector_type(4)))  unsigned int  u32x4;
typedef __attribute__((ext_vector_type(4)))  int           i32x4;
typedef __attribute__((ext_vector_type(8)))  int           i32x8;

#define EPSV 1e-5f

static __device__ __forceinline__ unsigned short f2bf_rn(float f) {
  unsigned u = __float_as_uint(f);
  unsigned r = (u + 0x7FFFu + ((u >> 16) & 1u)) >> 16;
  return (unsigned short)r;
}
// monotonic float<->uint key for atomicMax-based segment max
static __device__ __forceinline__ unsigned fkey(float f) {
  unsigned i = __float_as_uint(f);
  return (i & 0x80000000u) ? ~i : (i | 0x80000000u);
}
static __device__ __forceinline__ float fdec(unsigned k) {
  unsigned i = (k & 0x80000000u) ? (k & 0x7FFFFFFFu) : ~k;
  return __uint_as_float(i);
}

// ---------------- utility kernels ----------------
__global__ void k_zero_f32(float* p, long n) {
  long i = (long)blockIdx.x * 256 + threadIdx.x;
  if (i < n) p[i] = 0.f;
}
__global__ void k_fill_u32(unsigned* p, unsigned v, long n) {
  long i = (long)blockIdx.x * 256 + threadIdx.x;
  if (i < n) p[i] = v;
}
__global__ void k_f2bf(const float* __restrict__ s, unsigned short* __restrict__ d, long n) {
  long i = (long)blockIdx.x * 256 + threadIdx.x;
  if (i < n) d[i] = f2bf_rn(s[i]);
}
__global__ void k_copyf(const float* __restrict__ s, float* __restrict__ d, int n) {
  int i = blockIdx.x * 256 + threadIdx.x;
  if (i < n) d[i] = s[i];
}
// W (K x Nw, row-major f32) -> Wt ((nOff+n) x K, row-major bf16)  [B operand, N-major]
__global__ void k_prep_wt(const float* __restrict__ W, unsigned short* __restrict__ Wt,
                          int K, int Nw, int nOff) {
  int t = blockIdx.x * 256 + threadIdx.x;
  if (t >= K * Nw) return;
  int n = t / K, k = t - n * K;
  Wt[(size_t)(nOff + n) * K + k] = f2bf_rn(W[(size_t)k * Nw + n]);
}

// ---------------- bf16 WMMA GEMM with TDM-staged B in LDS ----------------
// Block = 8 waves. The whole B panel (Ncols x K bf16, Ncols % 64 == 0, K % 32 == 0,
// Ncols*K*2 <= 64KB) is DMA'd to LDS once per block via the Tensor Data Mover,
// then each wave computes a 16x64 C strip (4 WMMAs per A fragment).
// nGroups = Ncols/64 waves cover one M tile; mtPerBlk = 8/nGroups.
__global__ void __launch_bounds__(256)
k_gemm_bf16(const unsigned short* __restrict__ A,
            const unsigned short* __restrict__ Bt,
            const float* __restrict__ bias,
            float* __restrict__ C,
            int M, int Ncols, int K, int nGroups) {
  __shared__ unsigned short ldsW[256 * 128];  // 64 KB max panel

#if __has_builtin(__builtin_amdgcn_tensor_load_to_lds)
  if (threadIdx.x < 32) {  // one wave issues the tensor DMA (EXEC-independent)
    unsigned lds_off = (unsigned)(unsigned long long)(&ldsW[0]);
    unsigned long long ga = (unsigned long long)Bt;
    unsigned rowDw = (unsigned)(K >> 1);            // dwords per row (data_size=4B)
    u32x4 g0; i32x8 g1;
    i32x4 g2 = {0, 0, 0, 0}, g3 = {0, 0, 0, 0};
    i32x8 g4 = {0, 0, 0, 0, 0, 0, 0, 0};
    g0[0] = 1u;                                            // count=1 (valid descriptor)
    g0[1] = lds_off;                                       // lds_addr
    g0[2] = (unsigned)(ga & 0xFFFFFFFFull);                // global_addr[31:0]
    g0[3] = (unsigned)((ga >> 32) & 0x01FFFFFFull) | (2u << 30);  // ga[56:32] | type=2
    g1[0] = (int)(2u << 16);                               // data_size=2 -> 4 bytes
    g1[1] = (int)(rowDw << 16);                            // tensor_dim0[15:0]
    g1[2] = (int)(((unsigned)Ncols) << 16);                // tensor_dim1[15:0]
    g1[3] = (int)(rowDw << 16);                            // tile_dim0
    g1[4] = (int)(unsigned)Ncols;                          // tile_dim1 (tile_dim2=0)
    g1[5] = (int)rowDw;                                    // tensor_dim0_stride[31:0]
    g1[6] = 0;
    g1[7] = 0;
    __builtin_amdgcn_tensor_load_to_lds(g0, g1, g2, g3, g4, 0);
    __builtin_amdgcn_s_wait_tensorcnt(0);
  }
#else
  {
    const int nch = (Ncols * K) >> 3;   // 16B chunks
    for (int t = threadIdx.x; t < nch; t += 256)
      *(v4u*)(ldsW + (size_t)t * 8) = *(const v4u*)(Bt + (size_t)t * 8);
  }
#endif
  __syncthreads();

  int wave = threadIdx.x >> 5;
  int lane = threadIdx.x & 31;
  int mtPerBlk = 8 / nGroups;
  int mt = blockIdx.x * mtPerBlk + wave / nGroups;
  int ng = wave - (wave / nGroups) * nGroups;
  int mbase = mt * 16;
  if (mbase >= M) return;  // after the only barrier: safe

  int arow = mbase + (lane & 15);
  int aoff = ((lane >> 4) << 3);           // A 16-bit layout: two runs of 8 K-values
  int brow = ng * 64 + (lane & 15);        // LDS row (== output column)
  int boff = ((lane >> 4) << 3);           // B 16-bit layout: two runs of 8 K-values

  v8f acc[4] = {};
  const unsigned short* ap = A + (size_t)arow * K + aoff;
  for (int k0 = 0; k0 < K; k0 += 32) {
    v4u a0 = *(const v4u*)(ap + k0);
    v4u a1 = *(const v4u*)(ap + k0 + 16);
    v8u av = {a0.x, a0.y, a0.z, a0.w, a1.x, a1.y, a1.z, a1.w};
    v16bf af = __builtin_bit_cast(v16bf, av);
#pragma unroll
    for (int nt = 0; nt < 4; ++nt) {
      const unsigned short* bp = ldsW + (size_t)(brow + nt * 16) * K + boff + k0;
      v4u b0 = *(const v4u*)(bp);
      v4u b1 = *(const v4u*)(bp + 16);
      v8u bv = {b0.x, b0.y, b0.z, b0.w, b1.x, b1.y, b1.z, b1.w};
      acc[nt] = __builtin_amdgcn_wmma_f32_16x16x32_bf16(
          false, af, false, __builtin_bit_cast(v16bf, bv),
          (short)0, acc[nt], false, false);
    }
  }
  int r0 = mbase + ((lane >> 4) << 3);
#pragma unroll
  for (int nt = 0; nt < 4; ++nt) {
    int cc = ng * 64 + nt * 16 + (lane & 15);
    float bb = bias ? bias[cc] : 0.f;
#pragma unroll
    for (int j = 0; j < 8; ++j)
      C[(size_t)(r0 + j) * Ncols + cc] = acc[nt][j] + bb;
  }
}

// ---------------- edge-degree / self-loop attr ----------------
__global__ void k_deg(const int* __restrict__ ei, const float* __restrict__ ea,
                      float* __restrict__ deg, float* __restrict__ lsum, int E) {
  int e = blockIdx.x * 256 + threadIdx.x;
  if (e >= E) return;
  int d = ei[E + e];
  atomicAdd(&deg[d], 1.f);
  atomicAdd(&lsum[d], ea[e]);
}
__global__ void k_loopattr(const float* __restrict__ deg, float* __restrict__ lsum, int N) {
  int i = blockIdx.x * 256 + threadIdx.x;
  if (i < N) lsum[i] = lsum[i] / fmaxf(deg[i], 1.f);
}
__global__ void k_gcnt(const int* __restrict__ batch, float* __restrict__ gcnt, int N) {
  int i = blockIdx.x * 256 + threadIdx.x;
  if (i < N) atomicAdd(&gcnt[batch[i]], 1.f);
}

// ---------------- GATv2 attention logits (wave per edge) ----------------
__global__ void k_att(const float* __restrict__ xlr, const float* __restrict__ We,
                      const float* __restrict__ att, const int* __restrict__ ei,
                      const float* __restrict__ ea, const float* __restrict__ loopa,
                      float* __restrict__ logits, unsigned* __restrict__ nmax,
                      int E, int N) {
  int gid = blockIdx.x * 8 + (threadIdx.x >> 5);
  int lane = threadIdx.x & 31;
  if (gid >= E + N) return;
  int s, d; float a;
  if (gid < E) { s = ei[gid]; d = ei[E + gid]; a = ea[gid]; }
  else         { s = gid - E; d = s;           a = loopa[s]; }
  const float* xs = xlr + (size_t)s * 256;        // xl[s]
  const float* xd = xlr + (size_t)d * 256 + 128;  // xr[d]
  float pr[4];
#pragma unroll
  for (int h = 0; h < 4; ++h) {
    int ch = h * 32 + lane;
    float v = xs[ch] + xd[ch] + We[ch] * a;
    v = (v > 0.f) ? v : 0.2f * v;                 // leaky_relu(0.2)
    pr[h] = v * att[ch];
  }
#pragma unroll
  for (int off = 16; off; off >>= 1) {
#pragma unroll
    for (int h = 0; h < 4; ++h) pr[h] += __shfl_xor(pr[h], off, 32);
  }
  if (lane < 4) {
    logits[(size_t)gid * 4 + lane] = pr[lane];
    atomicMax(&nmax[(size_t)d * 4 + lane], fkey(pr[lane]));
  }
}

// ---------------- softmax-weighted scatter (wave per edge) ----------------
__global__ void k_scatter(const float* __restrict__ xlr, const int* __restrict__ ei,
                          const float* __restrict__ logits, const unsigned* __restrict__ nmax,
                          float* __restrict__ alsum, float* __restrict__ outun,
                          int E, int N) {
  int gid = blockIdx.x * 8 + (threadIdx.x >> 5);
  int lane = threadIdx.x & 31;
  if (gid >= E + N) return;
  int s, d;
  if (gid < E) { s = ei[gid]; d = ei[E + gid]; }
  else         { s = gid - E; d = s; }
  float al[4];
#pragma unroll
  for (int h = 0; h < 4; ++h)
    al[h] = expf(logits[(size_t)gid * 4 + h] - fdec(nmax[(size_t)d * 4 + h]));
  if (lane < 4) atomicAdd(&alsum[(size_t)d * 4 + lane], al[lane]);
  const float* xs = xlr + (size_t)s * 256;
  float* od = outun + (size_t)d * 128;
#pragma unroll
  for (int h = 0; h < 4; ++h) {
    int ch = h * 32 + lane;
    atomicAdd(&od[ch], xs[ch] * al[h]);
  }
}

__global__ void k_node_final(float* __restrict__ h, const float* __restrict__ alsum,
                             const float* __restrict__ bias, int N) {
  long t = (long)blockIdx.x * 256 + threadIdx.x;
  if (t >= (long)N * 128) return;
  int n = (int)(t >> 7), ch = (int)(t & 127);
  h[t] = h[t] / alsum[(size_t)n * 4 + (ch >> 5)] + bias[ch];
}

// ---------------- GraphNorm (+fused ReLU) ----------------
__global__ void k_gsum(const float* __restrict__ h, const int* __restrict__ batch,
                       float* __restrict__ gsum, int N) {
  long t = (long)blockIdx.x * 256 + threadIdx.x;
  if (t >= (long)N * 128) return;
  int n = (int)(t >> 7), ch = (int)(t & 127);
  atomicAdd(&gsum[batch[n] * 128 + ch], h[t]);
}
__global__ void k_gvar(const float* __restrict__ h, const int* __restrict__ batch,
                       const float* __restrict__ gsum, const float* __restrict__ gcnt,
                       const float* __restrict__ ms, float* __restrict__ gvar, int N) {
  long t = (long)blockIdx.x * 256 + threadIdx.x;
  if (t >= (long)N * 128) return;
  int n = (int)(t >> 7), ch = (int)(t & 127);
  int b = batch[n];
  float cnt = fmaxf(gcnt[b], 1.f);
  float sub = h[t] - ms[ch] * (gsum[b * 128 + ch] / cnt);
  atomicAdd(&gvar[b * 128 + ch], sub * sub);
}
__global__ void k_gnorm(const float* __restrict__ hin, const int* __restrict__ batch,
                        const float* __restrict__ gsum, const float* __restrict__ gvar,
                        const float* __restrict__ gcnt, const float* __restrict__ ms,
                        const float* __restrict__ w, const float* __restrict__ bb,
                        float* __restrict__ hout, int N) {
  long t = (long)blockIdx.x * 256 + threadIdx.x;
  if (t >= (long)N * 128) return;
  int n = (int)(t >> 7), ch = (int)(t & 127);
  int b = batch[n];
  float cnt = fmaxf(gcnt[b], 1.f);
  float sub = hin[t] - ms[ch] * (gsum[b * 128 + ch] / cnt);
  float v = w[ch] * sub * rsqrtf(gvar[b * 128 + ch] / cnt + EPSV) + bb[ch];
  hout[t] = fmaxf(v, 0.f);   // relu after graph_norm
}

// ---------------- gate network + attention pooling ----------------
__global__ void k_gate(const float* __restrict__ hW1, const float* __restrict__ g,
                       const float* __restrict__ be, const float* __restrict__ rm,
                       const float* __restrict__ rv, const float* __restrict__ W2,
                       const float* __restrict__ b2, const int* __restrict__ batch,
                       float* __restrict__ gt, unsigned* __restrict__ gmax, int N) {
  int n = blockIdx.x * 8 + (threadIdx.x >> 5);
  int lane = threadIdx.x & 31;
  if (n >= N) return;
  float acc = 0.f;
#pragma unroll
  for (int jj = 0; jj < 2; ++jj) {
    int j = lane + jj * 32;
    float v = (hW1[(size_t)n * 64 + j] - rm[j]) * rsqrtf(rv[j] + EPSV) * g[j] + be[j];
    acc += fmaxf(v, 0.f) * W2[j];
  }
#pragma unroll
  for (int off = 16; off; off >>= 1) acc += __shfl_xor(acc, off, 32);
  if (lane == 0) {
    float val = acc + b2[0];
    gt[n] = val;
    atomicMax(&gmax[batch[n]], fkey(val));
  }
}
__global__ void k_pool(const float* __restrict__ h, const float* __restrict__ gt,
                       const int* __restrict__ batch, const unsigned* __restrict__ gmax,
                       float* __restrict__ gwsum, float* __restrict__ zun, int N) {
  int n = blockIdx.x * 8 + (threadIdx.x >> 5);
  int lane = threadIdx.x & 31;
  if (n >= N) return;
  int b = batch[n];
  float gw = expf(gt[n] - fdec(gmax[b]));
  if (lane == 0) atomicAdd(&gwsum[b], gw);
#pragma unroll
  for (int hh = 0; hh < 4; ++hh) {
    int ch = hh * 32 + lane;
    atomicAdd(&zun[b * 128 + ch], gw * h[(size_t)n * 128 + ch]);
  }
}
__global__ void k_zfinal(const float* __restrict__ zun, const float* __restrict__ gwsum,
                         float* __restrict__ z) {
  int t = blockIdx.x * 256 + threadIdx.x;
  if (t >= 64 * 128) return;
  z[t] = zun[t] / fmaxf(gwsum[t >> 7], 1e-16f);
}

// ---------------- tiny MLP head ----------------
__global__ void k_mlp(const float* __restrict__ zin, const float* __restrict__ W,
                      const float* __restrict__ g, const float* __restrict__ be,
                      const float* __restrict__ rm, const float* __restrict__ rv,
                      float* __restrict__ zout, int K, int Nout) {
  int t = blockIdx.x * 256 + threadIdx.x;
  if (t >= 64 * Nout) return;
  int gi = t / Nout, j = t - gi * Nout;
  float acc = 0.f;
  for (int k = 0; k < K; ++k) acc += zin[gi * K + k] * W[k * Nout + j];
  float v = (acc - rm[j]) * rsqrtf(rv[j] + EPSV) * g[j] + be[j];
  zout[t] = (v > 0.f) ? v : 0.2f * v;
}
__global__ void k_final(const float* __restrict__ z, const float* __restrict__ Wf,
                        const float* __restrict__ bf, float* __restrict__ out) {
  int t = blockIdx.x * 128 + threadIdx.x;
  if (t >= 128) return;
  int gi = t >> 1, o = t & 1;
  float acc = bf[o];
  for (int k = 0; k < 256; ++k) acc += z[gi * 256 + k] * Wf[k * 2 + o];
  out[t] = acc;
}

// ---------------- launcher ----------------
extern "C" void kernel_launch(void* const* d_in, const int* in_sizes, int n_in,
                              void* d_out, int out_size, void* d_ws, size_t ws_size,
                              hipStream_t stream) {
  (void)n_in; (void)out_size; (void)ws_size;
  const float* x     = (const float*)d_in[0];
  const float* eattr = (const float*)d_in[1];
  const int*   ei    = (const int*)d_in[2];
  const int*   batch = (const int*)d_in[3];
  // params flattened in setup_inputs() insertion order:
  // c1: 4 Wl 5 bl 6 Wr 7 br 8 We 9 att 10 bias ; c2: 11..17
  // gn1: 18 w 19 b 20 ms ; gn2: 21..23
  // gate: 24 W1 25 g 26 be 27 rm 28 rv 29 W2 30 b2
  // mlp0: 31 W 32 g 33 be 34 rm 35 rv ; mlp1: 36..40 ; 41 Wf 42 bf
  const float* P[64];
  for (int i = 0; i < 43; ++i) P[i] = (const float*)d_in[i];

  const int N = in_sizes[3];       // 50000
  const int E = in_sizes[1];       // 400000
  const int Etot = E + N;
  const int G = 64;

  // workspace layout
  size_t off = 0;
  auto alloc = [&](size_t bytes) { size_t c = off; off = (off + bytes + 255) & ~(size_t)255; return c; };
  char* ws = (char*)d_ws;
  unsigned short* xbf   = (unsigned short*)(ws + alloc((size_t)N * 128 * 2));
  unsigned short* wt    = (unsigned short*)(ws + alloc((size_t)256 * 128 * 2));
  float* biascat        = (float*)(ws + alloc(256 * 4));
  float* xlr            = (float*)(ws + alloc((size_t)N * 256 * 4));
  float* deg            = (float*)(ws + alloc((size_t)N * 4));
  float* loopa          = (float*)(ws + alloc((size_t)N * 4));
  float* logits         = (float*)(ws + alloc((size_t)Etot * 4 * 4));
  unsigned* nmax        = (unsigned*)(ws + alloc((size_t)N * 4 * 4));
  float* alsum          = (float*)(ws + alloc((size_t)N * 4 * 4));
  float* bufP           = (float*)(ws + alloc((size_t)N * 128 * 4)); // conv out / h2
  float* bufQ           = (float*)(ws + alloc((size_t)N * 128 * 4)); // h1 / hW1
  float* gcnt           = (float*)(ws + alloc(G * 4));
  float* gsum           = (float*)(ws + alloc(G * 128 * 4));
  float* gvar           = (float*)(ws + alloc(G * 128 * 4));
  unsigned* gmax        = (unsigned*)(ws + alloc(G * 4));
  float* gwsum          = (float*)(ws + alloc(G * 4));
  float* zun            = (float*)(ws + alloc(G * 128 * 4));
  float* zA             = (float*)(ws + alloc(G * 256 * 4));
  float* zB             = (float*)(ws + alloc(G * 256 * 4));
  float* gt             = (float*)(ws + alloc((size_t)N * 4));

  auto cdivl = [](long a, long b) { return (unsigned)((a + b - 1) / b); };
  const unsigned NEG_INF_KEY = 0x007FFFFFu; // fkey(-inf)
  long NHC = (long)N * 128;
  int mtiles = N / 16;

  // ---- degree / self-loop attr / graph counts (edge-structure only; once) ----
  k_zero_f32<<<cdivl(N, 256), 256, 0, stream>>>(deg, N);
  k_zero_f32<<<cdivl(N, 256), 256, 0, stream>>>(loopa, N);
  k_zero_f32<<<1, 256, 0, stream>>>(gcnt, G);
  k_deg<<<cdivl(E, 256), 256, 0, stream>>>(ei, eattr, deg, loopa, E);
  k_loopattr<<<cdivl(N, 256), 256, 0, stream>>>(deg, loopa, N);
  k_gcnt<<<cdivl(N, 256), 256, 0, stream>>>(batch, gcnt, N);

  // ---- two GATv2 conv layers ----
  const float* hin_f32 = x;  // layer input (f32, N x 128)
  for (int layer = 0; layer < 2; ++layer) {
    int pb = (layer == 0) ? 4 : 11;
    const float *Wl = P[pb], *bl = P[pb + 1], *Wr = P[pb + 2], *br = P[pb + 3];
    const float *We = P[pb + 4], *att = P[pb + 5], *bias = P[pb + 6];
    int gb = (layer == 0) ? 18 : 21;
    const float *gw = P[gb], *gbb = P[gb + 1], *gms = P[gb + 2];

    // projections: xlr[:, :128]=x@Wl+bl, xlr[:, 128:]=x@Wr+br  (bf16 WMMA, TDM-staged B)
    k_f2bf<<<cdivl(NHC, 256), 256, 0, stream>>>(hin_f32, xbf, NHC);
    k_prep_wt<<<cdivl(128 * 128, 256), 256, 0, stream>>>(Wl, wt, 128, 128, 0);
    k_prep_wt<<<cdivl(128 * 128, 256), 256, 0, stream>>>(Wr, wt, 128, 128, 128);
    k_copyf<<<1, 256, 0, stream>>>(bl, biascat, 128);
    k_copyf<<<1, 256, 0, stream>>>(br, biascat + 128, 128);
    {
      int nGroups = 4;                      // 256 cols / 64
      dim3 grid(cdivl(mtiles, 8 / nGroups)), blk(256);
      k_gemm_bf16<<<grid, blk, 0, stream>>>(xbf, wt, biascat, xlr, N, 256, 128, nGroups);
    }

    // attention + softmax + scatter
    k_fill_u32<<<cdivl((long)N * 4, 256), 256, 0, stream>>>(nmax, NEG_INF_KEY, (long)N * 4);
    k_zero_f32<<<cdivl((long)N * 4, 256), 256, 0, stream>>>(alsum, (long)N * 4);
    k_zero_f32<<<cdivl(NHC, 256), 256, 0, stream>>>(bufP, NHC);
    k_att<<<cdivl(Etot, 8), 256, 0, stream>>>(xlr, We, att, ei, eattr, loopa, logits, nmax, E, N);
    k_scatter<<<cdivl(Etot, 8), 256, 0, stream>>>(xlr, ei, logits, nmax, alsum, bufP, E, N);
    k_node_final<<<cdivl(NHC, 256), 256, 0, stream>>>(bufP, alsum, bias, N);

    // graph norm + relu
    k_zero_f32<<<cdivl(G * 128, 256), 256, 0, stream>>>(gsum, G * 128);
    k_zero_f32<<<cdivl(G * 128, 256), 256, 0, stream>>>(gvar, G * 128);
    k_gsum<<<cdivl(NHC, 256), 256, 0, stream>>>(bufP, batch, gsum, N);
    k_gvar<<<cdivl(NHC, 256), 256, 0, stream>>>(bufP, batch, gsum, gcnt, gms, gvar, N);
    float* hout = (layer == 0) ? bufQ : bufP;   // layer1 -> bufQ, layer2 in place
    k_gnorm<<<cdivl(NHC, 256), 256, 0, stream>>>(bufP, batch, gsum, gvar, gcnt, gms, gw, gbb, hout, N);
    hin_f32 = hout;
  }
  float* h2 = bufP;  // final node features (N x 128)

  // ---- global attention pooling ----
  k_f2bf<<<cdivl(NHC, 256), 256, 0, stream>>>(h2, xbf, NHC);
  k_prep_wt<<<cdivl(128 * 64, 256), 256, 0, stream>>>(P[24], wt, 128, 64, 0);
  {
    int nGroups = 1;                        // 64 cols
    dim3 grid(cdivl(mtiles, 8 / nGroups)), blk(256);
    k_gemm_bf16<<<grid, blk, 0, stream>>>(xbf, wt, nullptr, bufQ, N, 64, 128, nGroups); // hW1
  }
  k_fill_u32<<<1, 256, 0, stream>>>(gmax, NEG_INF_KEY, G);
  k_gate<<<cdivl(N, 8), 256, 0, stream>>>(bufQ, P[25], P[26], P[27], P[28], P[29], P[30],
                                          batch, gt, gmax, N);
  k_zero_f32<<<1, 256, 0, stream>>>(gwsum, G);
  k_zero_f32<<<cdivl(G * 128, 256), 256, 0, stream>>>(zun, G * 128);
  k_pool<<<cdivl(N, 8), 256, 0, stream>>>(h2, gt, batch, gmax, gwsum, zun, N);
  k_zfinal<<<cdivl(G * 128, 256), 256, 0, stream>>>(zun, gwsum, zA);

  // ---- MLP head + final linear ----
  k_mlp<<<cdivl(G * 256, 256), 256, 0, stream>>>(zA, P[31], P[32], P[33], P[34], P[35], zB, 128, 256);
  k_mlp<<<cdivl(G * 256, 256), 256, 0, stream>>>(zB, P[36], P[37], P[38], P[39], P[40], zA, 256, 256);
  k_final<<<1, 128, 0, stream>>>(zA, P[41], P[42], (float*)d_out);
}
